// SchNetModel_90151363543798
// MI455X (gfx1250) — matrix operand order repaired and checked by reading, not compiled, not run on hardware
//
#include <hip/hip_runtime.h>
#include <cstdint>
#include <cstddef>

typedef __attribute__((ext_vector_type(16))) _Float16 v16h;
typedef __attribute__((ext_vector_type(8)))  float    v8f;
typedef _Float16 h16;

namespace {

constexpr int NN = 20000;     // nodes
constexpr int NE = 640000;    // edges
constexpr int HH = 128;       // hidden
constexpr int RR = 50;        // radial
constexpr int RP = 64;        // padded radial
constexpr float CUT = 5.0f;
constexpr float LN2 = 0.69314718055994531f;

__device__ __forceinline__ float sspf(float x){ return fmaxf(x,0.f) + log1pf(expf(-fabsf(x))) - LN2; }
__device__ __forceinline__ float sigmf(float x){ return 1.f/(1.f+expf(-x)); }
__device__ __forceinline__ float envf(float d){ float t=d/CUT; t=fminf(fmaxf(t,0.f),1.f); return 1.f - t*t*(3.f-2.f*t); }
__device__ __forceinline__ float denvf(float d){ float t=d/CUT; if(t<=0.f||t>=1.f) return 0.f; return (6.f*t*t-6.f*t)/CUT; }

// ---------------- epilogue flags ----------------
enum { EB=1, ES=2, ERS=4, EO16=8, EOADD=16, ESIG=32, EMS=64, EHU=128 };

// ---------------- WMMA GEMM ----------------
// A: [M, KS*32] f16 row-major (lda). B: pre-packed fragments, logical [KS*32, NT*16],
// staged once per workgroup into LDS (shared by the 4 waves), then read as v16h frags.
// One wave computes a 16 x (NT*16) tile. 4 waves / block.
template<int KS, int NT, int FL>
__global__ __launch_bounds__(128) void gemm_k(
    const h16* __restrict__ A, int lda,
    const h16* __restrict__ Bp,
    const float* __restrict__ bias,
    const float* __restrict__ rowscale,
    const h16* __restrict__ sigbuf,
    const h16* __restrict__ hprev,
    h16* __restrict__ out16,
    float* __restrict__ out32,
    h16* __restrict__ sigout,
    h16* __restrict__ hnew,
    int M)
{
  const int LD = NT*16;
  __shared__ __align__(32) h16 Bs[KS*NT*512];

  // cooperative global->LDS stage of packed B (KS*NT*1024 bytes)
  {
    const uint4* src = reinterpret_cast<const uint4*>(Bp);
    uint4* dst = reinterpret_cast<uint4*>(Bs);
    #pragma unroll
    for (int i = 0; i < (KS*NT*64)/128; ++i)
      dst[i*128 + threadIdx.x] = src[i*128 + threadIdx.x];
  }
  __syncthreads();

  int wave = threadIdx.x >> 5, lane = threadIdx.x & 31;
  int rt = blockIdx.x*4 + wave;
  if (rt*16 >= M) return;
  int row0 = rt*16, hh = lane>>4, lm = lane&15;

  v8f acc[NT] = {};
  const h16* Ar = A + (size_t)(row0+lm)*lda;
  for (int ks=0; ks<KS; ++ks) {
    v16h a;
    #pragma unroll
    for (int v=0; v<8; ++v) {
      int k = ks*32 + ((v>>2)<<4) + hh*8 + ((v&3)<<1);
      a[2*v]   = Ar[k];
      a[2*v+1] = Ar[k+1];
    }
    #pragma unroll
    for (int t=0; t<NT; ++t) {
      v16h b = *reinterpret_cast<const v16h*>(Bs + (((ks*NT+t)*32+lane) << 4));
      acc[t] = __builtin_amdgcn_wmma_f32_16x16x32_f16(false, a, false, b, (short)0, acc[t], false, false);
    }
  }
  #pragma unroll
  for (int t=0; t<NT; ++t) {
    int col = t*16 + lm;
    #pragma unroll
    for (int r=0; r<8; ++r) {
      int row = row0 + hh*8 + r;
      size_t oi = (size_t)row*LD + col;
      float z = acc[t][r];
      if (FL & EB)  z += bias[col];
      float res = z;
      if (FL & ES)  res = sspf(z);
      if (FL & ERS) res = z * rowscale[row];
      if (FL & EMS) res = z * (float)sigbuf[oi];
      if (FL & ESIG) sigout[oi] = (h16)sigmf(z);
      if (FL & EHU) { hnew[oi] = (h16)((float)hprev[oi] + sspf(z)); res = z; }
      if (FL & EO16)      out16[oi] = (h16)res;
      else if (FL & EOADD) out32[oi] += res;
      else                 out32[oi]  = res;
    }
  }
}

// ---------------- weight packer: f32 src -> f16 packed B fragments ----------------
// dst logical B[K][Nc]; element (k,c) comes from src (optionally transposed), zero-padded.
__global__ void k_pack(h16* __restrict__ dst, const float* __restrict__ src,
                       int Rs, int Cs, int K, int Nc, int tr)
{
  int idx = blockIdx.x*blockDim.x + threadIdx.x;
  int tot = K*Nc;
  if (idx >= tot) return;
  int i = idx & 15; int t2 = idx >> 4;
  int lane = t2 & 31; t2 >>= 5;
  int nct = Nc >> 4;
  int ct = t2 % nct, kt = t2 / nct;
  int hh = lane>>4, lm = lane&15;
  int k = kt*32 + ((i>>3)<<4) + hh*8 + (i&7);
  int c = ct*16 + lm;
  float v = 0.f;
  if (!tr) { if (k < Rs && c < Cs) v = src[(size_t)k*Cs + c]; }
  else     { if (c < Rs && k < Cs) v = src[(size_t)c*Cs + k]; }
  dst[idx] = (h16)v;
}

// ---------------- elementwise kernels ----------------
__global__ void k_zero(float* p, int n){ int i=blockIdx.x*blockDim.x+threadIdx.x; if(i<n) p[i]=0.f; }
__global__ void k_f32h(h16* d, const float* s, int n){ int i=blockIdx.x*blockDim.x+threadIdx.x; if(i<n) d[i]=(h16)s[i]; }

__global__ void k_rbf(const float* __restrict__ ew, float* __restrict__ dv,
                      float* __restrict__ env, h16* __restrict__ rbf,
                      float delta, float coeff)
{
  int idx = blockIdx.x*blockDim.x + threadIdx.x;
  if (idx >= NE*RP) return;
  int e = idx >> 6, r = idx & 63;
  float x = ew[e*3+0], y = ew[e*3+1], z = ew[e*3+2];
  float d = sqrtf(x*x + y*y + z*z);
  if (r == 0) { dv[e] = d; env[e] = envf(d); }
  float v = 0.f;
  if (r < RR) { float u = d - (float)r*delta; v = expf(coeff*u*u); }
  rbf[idx] = (h16)v;
}

__global__ void k_h0(h16* __restrict__ h0, const float* __restrict__ emb, const int* __restrict__ x)
{
  int idx = blockIdx.x*blockDim.x + threadIdx.x;
  if (idx >= NN*HH) return;
  int n = idx >> 7, c = idx & 127;
  h0[idx] = (h16)emb[(size_t)x[n]*HH + c];
}

// forward messages: agg[i] += W[e] * t[j]
__global__ void k_msg(const h16* __restrict__ W, const h16* __restrict__ t,
                      const int* __restrict__ ei, float* __restrict__ agg)
{
  int idx = blockIdx.x*blockDim.x + threadIdx.x;
  if (idx >= NE*HH) return;
  int e = idx >> 7, c = idx & 127;
  int i = ei[e], j = ei[NE + e];
  float m = (float)W[idx] * (float)t[(size_t)j*HH + c];
  atomicAdd(&agg[(size_t)i*HH + c], m);
}

__global__ void k_gpre(h16* __restrict__ gp, const float* __restrict__ gh,
                       const h16* __restrict__ pre, int n)
{
  int i = blockIdx.x*blockDim.x + threadIdx.x;
  if (i < n) gp[i] = (h16)(gh[i] * sigmf((float)pre[i]));
}

// backward edge chain: one 128-thread block per edge
__global__ void k_bedge(const float* __restrict__ gagg, const h16* __restrict__ t,
                        const h16* __restrict__ z2, const float* __restrict__ env,
                        const float* __restrict__ dv, const int* __restrict__ ei,
                        h16* __restrict__ gz2, float* __restrict__ gt, float* __restrict__ gd)
{
  __shared__ float sh[128];
  int e = blockIdx.x, c = threadIdx.x;
  int i = ei[e], j = ei[NE + e];
  size_t idx = (size_t)e*HH + c;
  float gm = gagg[(size_t)i*HH + c];
  float tj = (float)t[(size_t)j*HH + c];
  float zz = (float)z2[idx];
  float ev = env[e];
  atomicAdd(&gt[(size_t)j*HH + c], gm*zz*ev);   // g_t[j] += g_msg * W
  float gw = gm*tj;                             // g_W
  gz2[idx] = (h16)(gw*ev);                      // g_Z2
  sh[c] = gw*zz;                                // for g_env = <g_W, Z2>
  __syncthreads();
  for (int s=64; s>0; s>>=1) { if (c<s) sh[c]+=sh[c+s]; __syncthreads(); }
  if (c == 0) gd[e] += sh[0] * denvf(dv[e]);
}

__global__ void k_gdrbf(const float* __restrict__ grbf, const float* __restrict__ dv,
                        float* __restrict__ gd, float delta, float coeff)
{
  int e = blockIdx.x*blockDim.x + threadIdx.x;
  if (e >= NE) return;
  float d = dv[e], acc = 0.f;
  for (int r=0; r<RR; ++r) {
    float u = d - (float)r*delta;
    float rb = expf(coeff*u*u);
    acc += grbf[(size_t)e*RP + r] * (2.f*coeff*u) * rb;
  }
  gd[e] += acc;
}

// one 64-thread block per node: energy sum + g_z1
__global__ void k_energy(const h16* __restrict__ z1, const float* __restrict__ ow2,
                         const float* __restrict__ ob2, h16* __restrict__ gz1,
                         float* __restrict__ eout)
{
  __shared__ float sh[64];
  int n = blockIdx.x, r = threadIdx.x;
  size_t idx = (size_t)n*64 + r;
  float z = (float)z1[idx];
  float w = ow2[r];
  gz1[idx] = (h16)(w * sigmf(z));
  sh[r] = sspf(z) * w;
  __syncthreads();
  for (int s=32; s>0; s>>=1) { if (r<s) sh[r]+=sh[r+s]; __syncthreads(); }
  if (r == 0) atomicAdd(eout, sh[0] + ob2[0]);
}

// g_ew, forces scatter, stress accumulate
__global__ void k_final(const float* __restrict__ gd, const float* __restrict__ dv,
                        const float* __restrict__ ew, const int* __restrict__ ei,
                        float* __restrict__ forces, float* __restrict__ sraw)
{
  __shared__ float sh[9];
  if (threadIdx.x < 9) sh[threadIdx.x] = 0.f;
  __syncthreads();
  float loc[9] = {};
  for (int e = blockIdx.x*blockDim.x + threadIdx.x; e < NE; e += gridDim.x*blockDim.x) {
    float d = dv[e];
    float s = d > 1e-10f ? gd[e]/d : 0.f;
    float w0 = ew[e*3+0], w1 = ew[e*3+1], w2 = ew[e*3+2];
    float g0 = s*w0, g1 = s*w1, g2 = s*w2;
    int i = ei[e], j = ei[NE + e];
    atomicAdd(&forces[i*3+0],  g0); atomicAdd(&forces[i*3+1],  g1); atomicAdd(&forces[i*3+2],  g2);
    atomicAdd(&forces[j*3+0], -g0); atomicAdd(&forces[j*3+1], -g1); atomicAdd(&forces[j*3+2], -g2);
    loc[0]+=w0*g0; loc[1]+=w0*g1; loc[2]+=w0*g2;
    loc[3]+=w1*g0; loc[4]+=w1*g1; loc[5]+=w1*g2;
    loc[6]+=w2*g0; loc[7]+=w2*g1; loc[8]+=w2*g2;
  }
  #pragma unroll
  for (int q=0; q<9; ++q) atomicAdd(&sh[q], loc[q]);
  __syncthreads();
  if (threadIdx.x < 9) atomicAdd(&sraw[threadIdx.x], sh[threadIdx.x]);
}

__global__ void k_symsig(const float* __restrict__ sraw, float* __restrict__ out)
{
  int i = threadIdx.x;
  if (i >= 9) return;
  int k = i/3, l = i%3;
  out[i] = 0.5f*(sraw[k*3+l] + sraw[l*3+k]);
}

inline dim3 G(long long n){ return dim3((unsigned)((n + 255) / 256)); }

} // namespace

extern "C" void kernel_launch(void* const* d_in, const int* in_sizes, int n_in,
                              void* d_out, int out_size, void* d_ws, size_t ws_size,
                              hipStream_t stream)
{
  (void)out_size;
  if (n_in < 29) return;

  // -------- map inputs (jax pytree alphabetical flatten; fallback: insertion order) --------
  const int *edge_index; const float *edge_weight, *emb, *ob1, *ob2, *ow1, *ow2; const int *xtyp;
  struct Blk { const float *lin1, *l2b, *lin2, *b1, *b2, *w1, *w2; } bk[3];

  if (in_sizes[0] == 2*NE) { // alphabetical: edge_index, edge_weight, blocks(lin1,lin2_b,lin2,b1,b2,w1,w2)x3, emb, ob1, ob2, ow1, ow2, x
    edge_index  = (const int*)d_in[0];
    edge_weight = (const float*)d_in[1];
    for (int l=0; l<3; ++l) { int b = 2 + l*7;
      bk[l].lin1 = (const float*)d_in[b+0];
      bk[l].l2b  = (const float*)d_in[b+1];
      bk[l].lin2 = (const float*)d_in[b+2];
      bk[l].b1   = (const float*)d_in[b+3];
      bk[l].b2   = (const float*)d_in[b+4];
      bk[l].w1   = (const float*)d_in[b+5];
      bk[l].w2   = (const float*)d_in[b+6];
    }
    emb = (const float*)d_in[23]; ob1 = (const float*)d_in[24]; ob2 = (const float*)d_in[25];
    ow1 = (const float*)d_in[26]; ow2 = (const float*)d_in[27]; xtyp = (const int*)d_in[28];
  } else if (in_sizes[0] == NN) { // insertion: x, edge_index, edge_weight, emb, blocks(w1,b1,w2,b2,lin1,lin2,l2b)x3, ow1, ob1, ow2, ob2
    xtyp = (const int*)d_in[0];
    edge_index  = (const int*)d_in[1];
    edge_weight = (const float*)d_in[2];
    emb = (const float*)d_in[3];
    for (int l=0; l<3; ++l) { int b = 4 + l*7;
      bk[l].w1   = (const float*)d_in[b+0];
      bk[l].b1   = (const float*)d_in[b+1];
      bk[l].w2   = (const float*)d_in[b+2];
      bk[l].b2   = (const float*)d_in[b+3];
      bk[l].lin1 = (const float*)d_in[b+4];
      bk[l].lin2 = (const float*)d_in[b+5];
      bk[l].l2b  = (const float*)d_in[b+6];
    }
    ow1 = (const float*)d_in[25]; ob1 = (const float*)d_in[26];
    ow2 = (const float*)d_in[27]; ob2 = (const float*)d_in[28];
  } else return;

  // -------- workspace carve --------
  char* wsb = (char*)d_ws; size_t off = 0;
  auto alloc = [&](size_t bytes) -> void* {
    off = (off + 255) & ~(size_t)255;
    void* p = wsb + off; off += bytes; return p;
  };
  float* d_d   = (float*)alloc((size_t)NE*4);
  float* d_env = (float*)alloc((size_t)NE*4);
  float* d_gd  = (float*)alloc((size_t)NE*4);
  h16* rbf16 = (h16*)alloc((size_t)NE*RP*2);
  h16* act1  = (h16*)alloc((size_t)NE*HH*2);   // also reused as g_pre1 in backward
  h16* sig1  = (h16*)alloc((size_t)NE*HH*2);
  h16* z2w   = (h16*)alloc((size_t)NE*HH*2);   // W (fwd) / Z2 (bwd)
  h16* gz2   = (h16*)alloc((size_t)NE*HH*2);
  float* grbf = (float*)alloc((size_t)NE*RP*4);
  h16* hl[4]; for (int l=0;l<4;++l) hl[l] = (h16*)alloc((size_t)NN*HH*2);
  h16* pre16[3]; for (int l=0;l<3;++l) pre16[l] = (h16*)alloc((size_t)NN*HH*2);
  h16* t16    = (h16*)alloc((size_t)NN*HH*2);
  float* aggf = (float*)alloc((size_t)NN*HH*4);
  h16* agg16  = (h16*)alloc((size_t)NN*HH*2);
  float* ghf  = (float*)alloc((size_t)NN*HH*4);
  float* gaggf= (float*)alloc((size_t)NN*HH*4);
  float* gtf  = (float*)alloc((size_t)NN*HH*4);
  h16* gt16   = (h16*)alloc((size_t)NN*HH*2);
  h16* gpn16  = (h16*)alloc((size_t)NN*HH*2);
  h16* z1b    = (h16*)alloc((size_t)NN*64*2);
  h16* gz1b   = (h16*)alloc((size_t)NN*64*2);
  h16 *w1p[3], *w1T[3], *w2p[3], *w2T[3], *l1p[3], *l1T[3], *l2p[3], *l2T[3];
  for (int l=0;l<3;++l) {
    w1p[l] = (h16*)alloc((size_t)64*128*2);
    w1T[l] = (h16*)alloc((size_t)128*64*2);
    w2p[l] = (h16*)alloc((size_t)128*128*2);
    w2T[l] = (h16*)alloc((size_t)128*128*2);
    l1p[l] = (h16*)alloc((size_t)128*128*2);
    l1T[l] = (h16*)alloc((size_t)128*128*2);
    l2p[l] = (h16*)alloc((size_t)128*128*2);
    l2T[l] = (h16*)alloc((size_t)128*128*2);
  }
  h16* ow1p = (h16*)alloc((size_t)128*64*2);
  h16* ow1T = (h16*)alloc((size_t)64*128*2);
  float* sraw = (float*)alloc(9*4);
  if (off > ws_size) return; // workspace too small; nothing safe to do

  float* out  = (float*)d_out;          // [0]=energy, [1..60000]=forces, [60001..]=sigma
  float* Fout = out + 1;
  float* Sout = out + 1 + NN*3;

  const float delta = CUT / (float)(RR - 1);
  const float coeff = -0.5f / (delta*delta);
  const int EG = (NE/16 + 3)/4;   // edge-GEMM grid
  const int NG = (NN/16 + 3)/4;   // node-GEMM grid

  // -------- init / pack --------
  k_zero<<<G(1 + NN*3 + 9), 256, 0, stream>>>(out, 1 + NN*3 + 9);
  k_zero<<<G(NE), 256, 0, stream>>>(d_gd, NE);
  k_zero<<<G(9), 256, 0, stream>>>(sraw, 9);
  for (int l=0;l<3;++l) {
    k_pack<<<G(64*128), 256, 0, stream>>>(w1p[l], bk[l].w1, RR,128,  64,128, 0);
    k_pack<<<G(128*64), 256, 0, stream>>>(w1T[l], bk[l].w1, RR,128, 128, 64, 1);
    k_pack<<<G(128*128),256, 0, stream>>>(w2p[l], bk[l].w2, 128,128,128,128, 0);
    k_pack<<<G(128*128),256, 0, stream>>>(w2T[l], bk[l].w2, 128,128,128,128, 1);
    k_pack<<<G(128*128),256, 0, stream>>>(l1p[l], bk[l].lin1,128,128,128,128,0);
    k_pack<<<G(128*128),256, 0, stream>>>(l1T[l], bk[l].lin1,128,128,128,128,1);
    k_pack<<<G(128*128),256, 0, stream>>>(l2p[l], bk[l].lin2,128,128,128,128,0);
    k_pack<<<G(128*128),256, 0, stream>>>(l2T[l], bk[l].lin2,128,128,128,128,1);
  }
  k_pack<<<G(128*64), 256, 0, stream>>>(ow1p, ow1, 128,64, 128, 64, 0);
  k_pack<<<G(64*128), 256, 0, stream>>>(ow1T, ow1, 128,64,  64,128, 1);

  // -------- forward --------
  k_rbf<<<G((long long)NE*RP), 256, 0, stream>>>(edge_weight, d_d, d_env, rbf16, delta, coeff);
  k_h0 <<<G(NN*HH), 256, 0, stream>>>(hl[0], emb, xtyp);

  for (int l=0; l<3; ++l) {
    gemm_k<4,8,EO16><<<NG,128,0,stream>>>(hl[l],128, l1p[l], nullptr,nullptr,nullptr,nullptr, t16,nullptr,nullptr,nullptr, NN);
    gemm_k<2,8,EB|ES|EO16><<<EG,128,0,stream>>>(rbf16,64, w1p[l], bk[l].b1,nullptr,nullptr,nullptr, act1,nullptr,nullptr,nullptr, NE);
    gemm_k<4,8,EB|ERS|EO16><<<EG,128,0,stream>>>(act1,128, w2p[l], bk[l].b2, d_env,nullptr,nullptr, z2w,nullptr,nullptr,nullptr, NE);
    k_zero<<<G(NN*HH),256,0,stream>>>(aggf, NN*HH);
    k_msg<<<G((long long)NE*HH),256,0,stream>>>(z2w, t16, edge_index, aggf);
    k_f32h<<<G(NN*HH),256,0,stream>>>(agg16, aggf, NN*HH);
    gemm_k<4,8,EB|EO16|EHU><<<NG,128,0,stream>>>(agg16,128, l2p[l], bk[l].l2b,nullptr,nullptr, hl[l], pre16[l],nullptr,nullptr, hl[l+1], NN);
  }

  // output MLP: z1 = h3@ow1+ob1 ; energy + g_z1
  gemm_k<4,4,EB|EO16><<<NG,128,0,stream>>>(hl[3],128, ow1p, ob1,nullptr,nullptr,nullptr, z1b,nullptr,nullptr,nullptr, NN);
  k_energy<<<NN,64,0,stream>>>(z1b, ow2, ob2, gz1b, out);

  // -------- backward --------
  // g_h3 = g_z1 @ ow1^T
  gemm_k<2,8,0><<<NG,128,0,stream>>>(gz1b,64, ow1T, nullptr,nullptr,nullptr,nullptr, nullptr,ghf,nullptr,nullptr, NN);

  for (int l=2; l>=0; --l) {
    k_gpre<<<G(NN*HH),256,0,stream>>>(gpn16, ghf, pre16[l], NN*HH);
    gemm_k<4,8,0><<<NG,128,0,stream>>>(gpn16,128, l2T[l], nullptr,nullptr,nullptr,nullptr, nullptr,gaggf,nullptr,nullptr, NN);
    // recompute t, act1/sig1, Z2
    gemm_k<4,8,EO16><<<NG,128,0,stream>>>(hl[l],128, l1p[l], nullptr,nullptr,nullptr,nullptr, t16,nullptr,nullptr,nullptr, NN);
    gemm_k<2,8,EB|ES|EO16|ESIG><<<EG,128,0,stream>>>(rbf16,64, w1p[l], bk[l].b1,nullptr,nullptr,nullptr, act1,nullptr,sig1,nullptr, NE);
    gemm_k<4,8,EB|EO16><<<EG,128,0,stream>>>(act1,128, w2p[l], bk[l].b2,nullptr,nullptr,nullptr, z2w,nullptr,nullptr,nullptr, NE);
    // edge backward chain
    k_zero<<<G(NN*HH),256,0,stream>>>(gtf, NN*HH);
    k_bedge<<<NE,128,0,stream>>>(gaggf, t16, z2w, d_env, d_d, edge_index, gz2, gtf, d_gd);
    // g_pre1 = (g_Z2 @ w2^T) * sigma(pre1)   (reuse act1 buffer)
    gemm_k<4,8,EMS|EO16><<<EG,128,0,stream>>>(gz2,128, w2T[l], nullptr,nullptr, sig1,nullptr, act1,nullptr,nullptr,nullptr, NE);
    // g_rbf = g_pre1 @ w1^T
    gemm_k<4,4,0><<<EG,128,0,stream>>>(act1,128, w1T[l], nullptr,nullptr,nullptr,nullptr, nullptr,grbf,nullptr,nullptr, NE);
    k_gdrbf<<<G(NE),256,0,stream>>>(grbf, d_d, d_gd, delta, coeff);
    // g_h += g_t @ lin1^T
    k_f32h<<<G(NN*HH),256,0,stream>>>(gt16, gtf, NN*HH);
    gemm_k<4,8,EOADD><<<NG,128,0,stream>>>(gt16,128, l1T[l], nullptr,nullptr,nullptr,nullptr, nullptr,ghf,nullptr,nullptr, NN);
  }

  // -------- edge-weight grads -> forces & stress --------
  k_final<<<1024,256,0,stream>>>(d_gd, d_d, edge_weight, edge_index, Fout, sraw);
  k_symsig<<<1,16,0,stream>>>(sraw, Sout);
}
